// DDGCNet2_26242250179302
// MI455X (gfx1250) — matrix operand order repaired and checked by reading, not compiled, not run on hardware
//
// DDGCNet2 forward for MI455X (gfx1250, wave32, WMMA f16 16x16x32 / f32 accum).
// All GEMM-shaped work (point MLPs, STN, edge-conv MLPs, distance Gram) lowers
// to v_wmma_f32_16x16x32_f16. kNN top-k is fused with the WMMA distance tiles
// (per-row sorted lists resident in CDNA5's 320KB LDS; N x N never hits HBM).
// Round-3 upgrades: staging is now predication-free. Out-of-range rows/cols are
// loaded from clamped addresses WITHOUT selects (their garbage only reaches
// store-masked outputs); only the K-tail is zeroed, under a scalar-uniform
// branch. The kNN kernel reads from a pre-padded f16 copy of the feature slice
// so every fragment is two unconditional global_load_b128 (no cvt, no guards
// in the 500-tile streaming loop).
// NOTE: workspace layout assumes ws_size >= ~220 MB.

#include <hip/hip_runtime.h>
#include <hip/hip_bf16.h>

typedef __attribute__((ext_vector_type(16))) _Float16 v16h;
typedef __attribute__((ext_vector_type(8)))  _Float16 v8h;
typedef __attribute__((ext_vector_type(4)))  _Float16 v4h;
typedef __attribute__((ext_vector_type(2)))  _Float16 v2h;
typedef __attribute__((ext_vector_type(8)))  float    v8f;

#define NPTS 8000

// ---------------------------------------------------------------------------
// Fragment helpers (16x16x32 f16, CDNA5 ISA 7.12.2).
// A fragment from a row-major K-run: lane l (row m=l%16, half8=8*(l>=16)) is
// two contiguous 8-half runs at K = 2*half8+[0..7] and K = 2*half8+8+[0..7].
// B fragments: 16 contiguous halves per lane (pre-staged fragment order, or
// 16 contiguous channels starting at khalf for the Gram kernel).
// ---------------------------------------------------------------------------
__device__ static inline v16h ld_frag16(const _Float16* p) {
  v8h lo = *(const v8h*)p;
  v8h hi = *(const v8h*)(p + 8);
  return __builtin_shufflevector(lo, hi, 0, 1, 2, 3, 4, 5, 6, 7,
                                 8, 9, 10, 11, 12, 13, 14, 15);
}

__device__ static inline v16h ld_frag_a(const _Float16* p, int lane) {
  int half8 = (lane >= 16) ? 8 : 0;
  v8h lo = *(const v8h*)(p + 2 * half8);        // K = 2*half8+0..7
  v8h hi = *(const v8h*)(p + 2 * half8 + 8);    // K = 2*half8+8..15
  return __builtin_shufflevector(lo, hi, 0, 1, 2, 3, 4, 5, 6, 7,
                                 8, 9, 10, 11, 12, 13, 14, 15);
}

__device__ static inline float apply_act(float v, int act) {
  if (act == 1) v = fmaxf(v, 0.f);
  else if (act == 2) v = (v > 0.f) ? v : 0.2f * v;
  else if (act == 3) v = 1.f / (1.f + __expf(-v));
  return v;
}

// ---------------------------------------------------------------------------
// Generic GEMM: C[M,N] = act(A[M,K] @ B[K,N] + bias), A row stride lda.
// Block: 256 threads = 8 waves, tile 128(M) x 32(N), K-step 32. Each wave owns
// one 16-row strip and both 16-col tiles (one A-fragment feeds 2 WMMAs).
// Staging: full K-chunks load unconditionally from row/col-clamped pointers
// (out-of-range garbage is store-masked); only the K-tail zero-pads, under a
// scalar-uniform branch.
// ---------------------------------------------------------------------------
#define GT_M 128
#define GT_N 32
#define GT_K 32

__global__ __launch_bounds__(256)
void gemm_kernel(const float* __restrict__ A, int lda,
                 const float* __restrict__ B,
                 const float* __restrict__ bias,
                 float* __restrict__ C, int ldc,
                 int M, int N, int K, int act) {
  __shared__ alignas(16) _Float16 As[GT_M * GT_K];
  __shared__ alignas(16) _Float16 BSw[2 * 32 * 16];  // fragment-order B tiles
  int tid = threadIdx.x;
  int lane = tid & 31, wave = tid >> 5;
  int bm = blockIdx.y, bn = blockIdx.x;
  int r = tid >> 1, c0 = (tid & 1) * 16;
  int row = bm * GT_M + r;
  const float* arow = A + (long)min(row, M - 1) * lda;
  int cl = tid & 31, k0 = (tid >> 5) * 4;
  int colc = min(bn * GT_N + cl, N - 1);
  int nt2 = cl >> 4, nsl = cl & 15, khb = (k0 >= 16) ? 1 : 0;
  _Float16* bdst = &BSw[(nt2 * 32 + nsl + khb * 16) * 16 + (k0 & 15)];
  v8f acc0 = {}, acc1 = {};
  for (int kb = 0; kb < K; kb += GT_K) {
    v8h av0, av1;
    v4h bv;
    if (kb + GT_K <= K) {  // scalar-uniform fast path: no guards at all
      const float* ap = arow + kb + c0;
#pragma unroll
      for (int u = 0; u < 8; ++u) av0[u] = (_Float16)ap[u];
#pragma unroll
      for (int u = 0; u < 8; ++u) av1[u] = (_Float16)ap[8 + u];
      const float* bp = B + (long)(kb + k0) * N + colc;
#pragma unroll
      for (int u = 0; u < 4; ++u) bv[u] = (_Float16)bp[(long)u * N];
    } else {  // K-tail (at most once per kernel): zero-pad K
#pragma unroll
      for (int u = 0; u < 8; ++u) {
        int kg = kb + c0 + u;
        float vv = arow[min(kg, K - 1)];
        av0[u] = (_Float16)((kg < K) ? vv : 0.f);
      }
#pragma unroll
      for (int u = 0; u < 8; ++u) {
        int kg = kb + c0 + 8 + u;
        float vv = arow[min(kg, K - 1)];
        av1[u] = (_Float16)((kg < K) ? vv : 0.f);
      }
#pragma unroll
      for (int u = 0; u < 4; ++u) {
        int kg = kb + k0 + u;
        float vv = B[(long)min(kg, K - 1) * N + colc];
        bv[u] = (_Float16)((kg < K) ? vv : 0.f);
      }
    }
    *(v8h*)&As[r * GT_K + c0] = av0;
    *(v8h*)&As[r * GT_K + c0 + 8] = av1;
    *(v4h*)bdst = bv;
    if (kb + GT_K < K && row < M)  // gfx1250 global_prefetch_b8
      __builtin_prefetch(&arow[kb + GT_K + c0], 0, 1);
    __syncthreads();
    v16h af = ld_frag_a(&As[(wave * 16 + (lane & 15)) * GT_K], lane);
    v16h b0 = ld_frag16(&BSw[(0 * 32 + lane) * 16]);
    v16h b1 = ld_frag16(&BSw[(1 * 32 + lane) * 16]);
    acc0 = __builtin_amdgcn_wmma_f32_16x16x32_f16(false, af, false, b0,
                                                  (short)0, acc0, false, false);
    acc1 = __builtin_amdgcn_wmma_f32_16x16x32_f16(false, af, false, b1,
                                                  (short)0, acc1, false, false);
    __syncthreads();
  }
  int half8 = (lane >= 16) ? 8 : 0;
  int ncol0 = bn * GT_N + (lane & 15);
  int ncol1 = ncol0 + 16;
#pragma unroll
  for (int t = 0; t < 8; ++t) {
    int orow = bm * GT_M + wave * 16 + t + half8;
    if (orow < M) {
      if (ncol0 < N) {
        float v = acc0[t];
        if (bias) v += bias[ncol0];
        C[(long)orow * ldc + ncol0] = apply_act(v, act);
      }
      if (ncol1 < N) {
        float v = acc1[t];
        if (bias) v += bias[ncol1];
        C[(long)orow * ldc + ncol1] = apply_act(v, act);
      }
    }
  }
}

// ---------------------------------------------------------------------------
// Zero-padded f16 copy of a feature slice (row stride lda -> dense ldp = 32*NCH)
// so the Gram/top-k kernel streams unconditional global_load_b128 fragments.
// ---------------------------------------------------------------------------
__global__ void cvt_pad_kernel(const float* X, int lda, int C,
                               _Float16* Xp, int ldp, int n) {
  int e = blockIdx.x * blockDim.x + threadIdx.x;
  if (e >= n * ldp) return;
  int i = e / ldp, c = e % ldp;
  float vv = X[(long)i * lda + min(c, C - 1)];
  Xp[e] = (_Float16)((c < C) ? vv : 0.f);
}

// ---------------------------------------------------------------------------
// Fused pairwise-distance + top-k over the padded f16 features. One wave owns
// 16 query rows; query fragments are hoisted (2x b128 each). Per column tile:
// per-lane B fragment = 16 contiguous padded channels (2x b128, unconditional),
// one WMMA per 32-channel chunk, d2 = |i|^2+|j|^2-2<i,j> (fp32 norms), then
// lanes 0..15 insertion-sort into per-row top-allk lists in LDS.
// ---------------------------------------------------------------------------
#define TOPK_MAX 544

template <int NCH>
__global__ __launch_bounds__(32)
void dist_topk_kernel(const _Float16* __restrict__ Xp,
                      const float* __restrict__ sqn,
                      int* __restrict__ top, int allk, int n) {
  const int LDP = NCH * 32;
  __shared__ float dv[16][TOPK_MAX];
  __shared__ int   di[16][TOPK_MAX];
  __shared__ float dt[16][16];
  int lane = threadIdx.x;
  int rowbase = blockIdx.x * 16;
  int r = lane & 15;
  int half8 = (lane >= 16) ? 8 : 0;
  int kh = (lane >= 16) ? 16 : 0;
  v16h afr[NCH];
  {
    const _Float16* prow = Xp + (long)(rowbase + (lane & 15)) * LDP;
#pragma unroll
    for (int kc = 0; kc < NCH; ++kc)
      afr[kc] = ld_frag_a(prow + kc * 32, lane);
  }
  int cnt = 0;
  int ntiles = n >> 4;
  for (int ct = 0; ct < ntiles; ++ct) {
    int colbase = ct * 16;
    const _Float16* pcol = Xp + (long)(colbase + (lane & 15)) * LDP + kh;
    v8f acc = {};
#pragma unroll
    for (int kc = 0; kc < NCH; ++kc) {
      v16h bf = ld_frag16(pcol + kc * 32);
      acc = __builtin_amdgcn_wmma_f32_16x16x32_f16(false, afr[kc], false, bf,
                                                   (short)0, acc, false, false);
    }
    float scol = sqn[colbase + (lane & 15)];
#pragma unroll
    for (int t = 0; t < 8; ++t) {
      int mm = t + half8;
      float d = sqn[rowbase + mm] + scol - 2.f * acc[t];
      dt[mm][lane & 15] = (d < 0.f) ? 0.f : d;
    }
    __syncthreads();
    if (lane < 16) {
#pragma unroll 1
      for (int j = 0; j < 16; ++j) {
        float cv = dt[r][j];
        int gcol = colbase + j;
        if (cnt < allk) {
          int p = cnt++;
          while (p > 0 && dv[r][p - 1] > cv) {
            dv[r][p] = dv[r][p - 1]; di[r][p] = di[r][p - 1]; --p;
          }
          dv[r][p] = cv; di[r][p] = gcol;
        } else if (cv < dv[r][allk - 1]) {
          int p = allk - 1;
          while (p > 0 && dv[r][p - 1] > cv) {
            dv[r][p] = dv[r][p - 1]; di[r][p] = di[r][p - 1]; --p;
          }
          dv[r][p] = cv; di[r][p] = gcol;
        }
      }
    }
    __syncthreads();
  }
  if (lane < 16)
    for (int t = 0; t < allk; ++t)
      top[(long)(rowbase + r) * allk + t] = di[r][t];
}

// ---------------------------------------------------------------------------
// Fused edge-conv tail. Host decomposes the edge MLP:
//   relu([xi, xj-xi]@W1+b1) = relu(P_i + Q_j),  P = X@(W1a-W1b)+b1, Q = X@W1b.
// Per point i: A = relu(P_i + Q_idx[i][:]) (k x h f16 rows in LDS; rows j>=k
// hold garbage that the j<k max-reduce never reads), WMMA (k x h)@(h x o) with
// A-fragments hoisted across the o-tiles, column max over k neighbors + b2.
// Block = 128 threads = 4 waves = 4 points. W2 staged fragment-ordered with a
// uniform k-quad loop (thread = column, column clamp without select).
// ---------------------------------------------------------------------------
#define EC_HP 128
#define EC_OP 128

template <int KCH>
__global__ __launch_bounds__(128)
void edge_conv_kernel(const float* __restrict__ P, const float* __restrict__ Q,
                      int hdim, const int* __restrict__ idx, int idxw, int k,
                      const float* __restrict__ W2,
                      const float* __restrict__ b2, int o,
                      float* __restrict__ out) {
  __shared__ alignas(16) _Float16 W2sw[8 * 4 * 32 * 16];  // [ntile][kch][lane][16]
  __shared__ alignas(16) _Float16 As[4][32 * EC_HP];
  __shared__ float Ds[4][32 * 16];
  int tid = threadIdx.x, lane = tid & 31, wave = tid >> 5;
  {  // stage W2 fragment-ordered: thread = column, serial k-quads (uniform k)
    int c = tid;
    int cc = min(c, o - 1);  // column clamp, garbage cols are store-masked
    int nt2 = c >> 4, nsl = c & 15;
    for (int kq = 0; kq < KCH * 8; ++kq) {
      int k0 = kq * 4;
      v4h wv;
      if (k0 + 4 <= hdim) {  // uniform (hdim % 4 == 0 in all configs)
#pragma unroll
        for (int u = 0; u < 4; ++u) wv[u] = (_Float16)W2[(long)(k0 + u) * o + cc];
      } else {
#pragma unroll
        for (int u = 0; u < 4; ++u) {
          float vv = W2[(long)min(k0 + u, hdim - 1) * o + cc];
          wv[u] = (_Float16)((k0 + u < hdim) ? vv : 0.f);
        }
      }
      int kch = k0 >> 5, khf = ((k0 >> 4) & 1);
      *(v4h*)&W2sw[(((nt2 * 4) + kch) * 32 + nsl + khf * 16) * 16 + (k0 & 15)] = wv;
    }
  }
  int i = blockIdx.x * 4 + wave;
  {  // build A rows: relu(P_i + Q_j); rows j>=k = garbage (never max-reduced)
    int j = lane;
    int nb = idx[(long)i * idxw + min(j, k - 1)];
    const float* Pi = P + (long)i * hdim;
    const float* Qn = Q + (long)nb * hdim;
    int c = 0;
    for (; c + 2 <= hdim; c += 2) {  // hdim is even in all configs
      v2h t;
      t[0] = (_Float16)fmaxf(Pi[c] + Qn[c], 0.f);
      t[1] = (_Float16)fmaxf(Pi[c + 1] + Qn[c + 1], 0.f);
      *(v2h*)&As[wave][j * EC_HP + c] = t;
    }
    for (; c < KCH * 32; c += 2) {  // zero-pad channel tail
      v2h t = {};
      *(v2h*)&As[wave][j * EC_HP + c] = t;
    }
  }
  __syncthreads();
  // hoist A fragments (2 m-tiles x KCH chunks) into registers
  v16h afr[2][KCH];
#pragma unroll
  for (int mt = 0; mt < 2; ++mt)
#pragma unroll
    for (int kc = 0; kc < KCH; ++kc)
      afr[mt][kc] = ld_frag_a(&As[wave][(mt * 16 + (lane & 15)) * EC_HP + kc * 32], lane);
  int notile = (o + 15) >> 4;
  int half8 = (lane >= 16) ? 8 : 0;
  for (int nt = 0; nt < notile; ++nt) {
    v8f acc0 = {}, acc1 = {};
#pragma unroll
    for (int kc = 0; kc < KCH; ++kc) {
      v16h bf = ld_frag16(&W2sw[((nt * 4 + kc) * 32 + lane) * 16]);
      acc0 = __builtin_amdgcn_wmma_f32_16x16x32_f16(false, afr[0][kc], false, bf,
                                                    (short)0, acc0, false, false);
      acc1 = __builtin_amdgcn_wmma_f32_16x16x32_f16(false, afr[1][kc], false, bf,
                                                    (short)0, acc1, false, false);
    }
#pragma unroll
    for (int t = 0; t < 8; ++t) {
      Ds[wave][(t + half8) * 16 + (lane & 15)] = acc0[t];
      Ds[wave][(16 + t + half8) * 16 + (lane & 15)] = acc1[t];
    }
    __syncthreads();
    if (lane < 16) {
      int col = nt * 16 + lane;
      if (col < o) {
        float mx = -3.4e38f;
        for (int j = 0; j < k; ++j) mx = fmaxf(mx, Ds[wave][j * 16 + lane]);
        out[(long)i * o + col] = mx + b2[col];
      }
    }
    __syncthreads();
  }
}

// ------------------------- small support kernels ---------------------------
__global__ void sqnorm_kernel(const float* X, int lda, int C, float* sqn, int n) {
  int i = blockIdx.x * blockDim.x + threadIdx.x;
  if (i >= n) return;
  float s = 0.f;
  for (int c = 0; c < C; ++c) { float v = X[(long)i * lda + c]; s += v * v; }
  sqn[i] = s;
}

__global__ void slice_idx_kernel(const int* top, int topw, int* dst, int dstw,
                                 int off, int k, int stride) {
  int i = blockIdx.x, t = threadIdx.x;
  if (t < k) dst[(long)i * dstw + off + t] = top[(long)i * topw + t * stride];
}

__global__ void wdiff_kernel(const float* W1, int tot, float* Wd) {
  int e = blockIdx.x * blockDim.x + threadIdx.x;
  if (e < tot) Wd[e] = W1[e] - W1[tot + e];
}

__global__ void max2_lrelu_kernel(const float* a, const float* b, int w,
                                  float* dst, int dld, int doff, int n) {
  int e = blockIdx.x * blockDim.x + threadIdx.x;
  if (e >= n * w) return;
  int i = e / w, c = e % w;
  float v = fmaxf(a[(long)i * w + c], b[(long)i * w + c]);
  dst[(long)i * dld + doff + c] = (v > 0.f) ? v : 0.2f * v;
}

__global__ void colmax_kernel(const float* A, int K, int n, float* out) {
  int c = blockIdx.x * blockDim.x + threadIdx.x;
  if (c >= K) return;
  float m = -3.4e38f;
  for (int i = 0; i < n; ++i) m = fmaxf(m, A[(long)i * K + c]);
  out[c] = m;
}

__global__ void stn_apply_kernel(const float* X, const float* t36, float* x0, int n) {
  int i = blockIdx.x * blockDim.x + threadIdx.x;
  if (i >= n) return;
  float xi[6];
  for (int c = 0; c < 6; ++c) xi[c] = X[i * 6 + c];
  for (int d = 0; d < 6; ++d) {
    float s = 0.f;
    for (int c = 0; c < 6; ++c)
      s += xi[c] * (t36[c * 6 + d] + ((c == d) ? 1.f : 0.f));
    x0[i * 6 + d] = s;
  }
}

__device__ __constant__ int GROUP_LUT_DEV[16] =
    {0, 0, 0, 0, 1, 1, 2, 3, 3, 3, 3, 2, 1, 1, 0, 0};

__global__ void lut_kernel(const int* seg, int* gid, int n) {
  int i = blockIdx.x * blockDim.x + threadIdx.x;
  if (i < n) gid[i] = GROUP_LUT_DEV[seg[i] & 15];
}

__global__ void segmax_kernel(const float* h2, const int* gid, int n, float* g4) {
  int t = blockIdx.x * blockDim.x + threadIdx.x;
  if (t >= 4 * 128) return;
  int grp = t >> 7, c = t & 127;
  float m = -3.4e38f;
  for (int i = 0; i < n; ++i)
    if (gid[i] == grp) m = fmaxf(m, h2[(long)i * 128 + c]);
  g4[grp * 128 + c] = m;
}

__global__ void catgp_kernel(const float* h2, const int* gid, const float* g4,
                             float* f, int n) {
  int e = blockIdx.x * blockDim.x + threadIdx.x;
  if (e >= n * 256) return;
  int i = e >> 8, c = e & 255;
  f[e] = (c < 128) ? h2[(long)i * 128 + c] : g4[gid[i] * 128 + (c - 128)];
}

__global__ void rank_kernel(const int* gid, int* dstof, int n) {
  if (threadIdx.x != 0 || blockIdx.x != 0) return;
  int cnt[4] = {0, 0, 0, 0};
  for (int i = 0; i < n; ++i) cnt[gid[i]]++;
  int off[4]; off[0] = 0;
  for (int g = 1; g < 4; ++g) off[g] = off[g - 1] + cnt[g - 1];
  int seen[4] = {0, 0, 0, 0};
  for (int i = 0; i < n; ++i) { int g = gid[i]; dstof[i] = off[g] + seen[g]++; }
}

__global__ void permscatter_kernel(const float* pn3, const int* dstof,
                                   float* cat, int n) {
  int e = blockIdx.x * blockDim.x + threadIdx.x;
  if (e >= n * 256) return;
  int i = e >> 8, c = e & 255;
  cat[(long)dstof[i] * 656 + 400 + c] = pn3[e];
}

__global__ void mul_kernel(const float* a, int aw, const float* s, int sw,
                           float* dst, int dld, int doff, int w, int n) {
  int e = blockIdx.x * blockDim.x + threadIdx.x;
  if (e >= n * w) return;
  int i = e / w, c = e % w;
  dst[(long)i * dld + doff + c] = a[(long)i * aw + c] * s[(long)i * sw + c];
}

__global__ void add_kernel(const float* a, const float* b, float* y, long cnt) {
  long e = blockIdx.x * (long)blockDim.x + threadIdx.x;
  if (e < cnt) y[e] = a[e] + b[e];
}

// ------------------------------ host side ----------------------------------
static inline void launch_gemm(hipStream_t s, const float* A, int lda,
                               const float* B, const float* bias, float* C,
                               int ldc, int M, int N, int K, int act) {
  dim3 grid((N + GT_N - 1) / GT_N, (M + GT_M - 1) / GT_M);
  gemm_kernel<<<grid, 256, 0, s>>>(A, lda, B, bias, C, ldc, M, N, K, act);
}

extern "C" void kernel_launch(void* const* d_in, const int* in_sizes, int n_in,
                              void* d_out, int out_size, void* d_ws,
                              size_t ws_size, hipStream_t stream) {
  (void)in_sizes; (void)n_in; (void)out_size; (void)ws_size;
  // d_in: [x, pos, seg_group] then params leaves in jax sorted-key tree order:
  // d1..d3, e1..e3 (each a.{W1,W2,b1,b2}, b.{...}), fi, g1, gfi, mlp2, out,
  // pn.{l1,l2,l3}, rb1.{l1,l2,rs}, rb2.{...}, stn.{c1,c2,c3,f1,f2,f3};
  // per-layer leaves sorted W before b ('W' < 'b').
  const float* X   = (const float*)d_in[0];
  const float* POS = (const float*)d_in[1];
  const int*   SEG = (const int*)d_in[2];
  int p = 3;
  struct Branch { const float *W1, *W2, *b1, *b2; };
  struct Block { Branch a, b; };
  auto getBranch = [&]() {
    Branch br;
    br.W1 = (const float*)d_in[p++]; br.W2 = (const float*)d_in[p++];
    br.b1 = (const float*)d_in[p++]; br.b2 = (const float*)d_in[p++];
    return br;
  };
  auto getBlock = [&]() { Block bl; bl.a = getBranch(); bl.b = getBranch(); return bl; };
  Block D1b = getBlock(), D2b = getBlock(), D3b = getBlock();
  Block E1b = getBlock(), E2b = getBlock(), E3b = getBlock();
  struct Lin { const float *W, *b; };
  auto getLin = [&]() {
    Lin l; l.W = (const float*)d_in[p++]; l.b = (const float*)d_in[p++]; return l;
  };
  Lin FI = getLin(), G1 = getLin(), GFI = getLin(), MLP2 = getLin(), OUTL = getLin();
  Lin PN1 = getLin(), PN2 = getLin(), PN3 = getLin();
  Lin RB1L1 = getLin(), RB1L2 = getLin(), RB1RS = getLin();
  Lin RB2L1 = getLin(), RB2L2 = getLin(), RB2RS = getLin();
  Lin SC1 = getLin(), SC2 = getLin(), SC3 = getLin();
  Lin SF1 = getLin(), SF2 = getLin(), SF3 = getLin();

  // workspace layout (floats)
  float* ws = (float*)d_ws;
  long off = 0;
  auto alloc = [&](long cnt) { float* q = ws + off; off += cnt; return q; };
  float* feat = alloc((long)NPTS * 380);   // [x1 | x2 | x3] channel-concat
  float* x0   = alloc(NPTS * 6);
  float* sqn  = alloc(NPTS);
  float* P    = alloc(NPTS * 128);
  float* Q    = alloc(NPTS * 128);
  float* Wd   = alloc(256 * 128);
  float* ea   = alloc(NPTS * 128);
  float* eb   = alloc(NPTS * 128);
  float* bigA = alloc((long)NPTS * 1024);
  float* bigB = alloc((long)NPTS * 1024);
  float* cat  = alloc((long)NPTS * 656);   // [g | xg]
  float* r1   = alloc(NPTS * 512);
  float* r2   = alloc(NPTS * 512);
  float* r3   = alloc(NPTS * 512);
  float* vecs = alloc(4096);               // stn: g(1024), t512, t256, t36
  float* g4   = alloc(4 * 128);
  float* f256 = alloc(NPTS * 256);
  float* pn3  = alloc(NPTS * 256);
  _Float16* xpad = (_Float16*)alloc(NPTS * 64);  // padded f16 features (<=128ch)
  int* ftop  = (int*)alloc((long)NPTS * 540);
  int* ptop  = (int*)alloc((long)NPTS * 540);
  int* ia    = (int*)alloc(NPTS * 32);
  int* ib    = (int*)alloc(NPTS * 16);
  int* gid   = (int*)alloc(NPTS);
  int* dstof = (int*)alloc(NPTS);

  auto topk = [&](const float* Xp, int lda, int C, int* top, int allk) {
    sqnorm_kernel<<<(NPTS + 255) / 256, 256, 0, stream>>>(Xp, lda, C, sqn, NPTS);
    int nch = (C + 31) / 32;
    int ldp = nch * 32;
    cvt_pad_kernel<<<(NPTS * ldp + 255) / 256, 256, 0, stream>>>(
        Xp, lda, C, xpad, ldp, NPTS);
    if (nch == 1)
      dist_topk_kernel<1><<<NPTS / 16, 32, 0, stream>>>(xpad, sqn, top, allk, NPTS);
    else if (nch == 2)
      dist_topk_kernel<2><<<NPTS / 16, 32, 0, stream>>>(xpad, sqn, top, allk, NPTS);
    else
      dist_topk_kernel<4><<<NPTS / 16, 32, 0, stream>>>(xpad, sqn, top, allk, NPTS);
  };
  auto slice = [&](const int* top, int topw, int* dst, int dstw, int o2,
                   int kc, int stride) {
    slice_idx_kernel<<<NPTS, 32, 0, stream>>>(top, topw, dst, dstw, o2, kc, stride);
  };
  auto run_branch = [&](const float* xin, int lda, int cin, const Branch& br,
                        int h, int o, const int* idxp, int idxw, int kk,
                        float* outp) {
    int tot = cin * h;
    wdiff_kernel<<<(tot + 255) / 256, 256, 0, stream>>>(br.W1, tot, Wd);
    launch_gemm(stream, xin, lda, Wd, br.b1, P, h, NPTS, h, cin, 0);
    launch_gemm(stream, xin, lda, br.W1 + (long)cin * h, nullptr, Q, h,
                NPTS, h, cin, 0);
    int kch = (h + 31) / 32;
    switch (kch) {
      case 1: edge_conv_kernel<1><<<NPTS / 4, 128, 0, stream>>>(
                  P, Q, h, idxp, idxw, kk, br.W2, br.b2, o, outp); break;
      case 2: edge_conv_kernel<2><<<NPTS / 4, 128, 0, stream>>>(
                  P, Q, h, idxp, idxw, kk, br.W2, br.b2, o, outp); break;
      case 3: edge_conv_kernel<3><<<NPTS / 4, 128, 0, stream>>>(
                  P, Q, h, idxp, idxw, kk, br.W2, br.b2, o, outp); break;
      default: edge_conv_kernel<4><<<NPTS / 4, 128, 0, stream>>>(
                  P, Q, h, idxp, idxw, kk, br.W2, br.b2, o, outp); break;
    }
  };
  auto max2 = [&](int w, int foff) {
    max2_lrelu_kernel<<<(NPTS * w + 255) / 256, 256, 0, stream>>>(
        ea, eb, w, feat, 380, foff, NPTS);
  };

  // ---- STN ----
  launch_gemm(stream, X, 6, SC1.W, SC1.b, P, 64, NPTS, 64, 6, 1);
  launch_gemm(stream, P, 64, SC2.W, SC2.b, Q, 128, NPTS, 128, 64, 1);
  launch_gemm(stream, Q, 128, SC3.W, SC3.b, bigA, 1024, NPTS, 1024, 128, 1);
  colmax_kernel<<<4, 256, 0, stream>>>(bigA, 1024, NPTS, vecs);
  launch_gemm(stream, vecs, 1024, SF1.W, SF1.b, vecs + 1024, 512, 1, 512, 1024, 1);
  launch_gemm(stream, vecs + 1024, 512, SF2.W, SF2.b, vecs + 1536, 256, 1, 256, 512, 1);
  launch_gemm(stream, vecs + 1536, 256, SF3.W, SF3.b, vecs + 1792, 36, 1, 36, 256, 0);
  stn_apply_kernel<<<(NPTS + 255) / 256, 256, 0, stream>>>(X, vecs + 1792, x0, NPTS);

  // ---- positional kNN (top-540 sorted once, prefixes/strides reused) ----
  topk(POS, 3, 3, ptop, 540);

  // ---- stage 1: x0 (C=6) -> feat[:, 0:60) ----
  topk(x0, 6, 6, ftop, 240);
  slice(ftop, 240, ia, 32, 0, 32, 1);              // E1 ia: knn feat 32
  slice(ftop, 240, ib, 16, 0, 12, 1);              // E1 ib: knn feat 12
  run_branch(x0, 6, 6, E1b.a, 24, 24, ia, 32, 32, ea);
  run_branch(x0, 6, 6, E1b.b, 24, 24, ib, 16, 12, eb);
  max2(24, 0);
  slice(ftop, 240, ia, 32, 0, 32, 7);              // D1 ia: dknn 240/32
  slice(ftop, 240, ib, 16, 0, 12, 20);             // D1 ib: dknn 240/12
  run_branch(x0, 6, 6, D1b.a, 36, 36, ia, 32, 32, ea);
  run_branch(x0, 6, 6, D1b.b, 36, 36, ib, 16, 12, eb);
  max2(36, 24);

  // ---- stage 2: x1 = feat[:,0:60) -> feat[:,60:180) ----
  topk(feat, 380, 60, ftop, 360);
  slice(ftop, 360, ia, 32, 0, 16, 1);  slice(ptop, 540, ia, 32, 16, 16, 1);
  slice(ftop, 360, ib, 16, 0, 6, 1);   slice(ptop, 540, ib, 16, 6, 6, 1);
  run_branch(feat, 380, 60, E2b.a, 48, 48, ia, 32, 32, ea);
  run_branch(feat, 380, 60, E2b.b, 48, 48, ib, 16, 12, eb);
  max2(48, 60);
  slice(ftop, 360, ia, 32, 0, 18, 20); slice(ptop, 540, ia, 32, 18, 14, 25);
  slice(ftop, 360, ib, 16, 0, 4, 90);  slice(ptop, 540, ib, 16, 4, 6, 60);
  run_branch(feat, 380, 60, D2b.a, 72, 72, ia, 32, 32, ea);
  run_branch(feat, 380, 60, D2b.b, 72, 72, ib, 16, 10, eb);
  max2(72, 108);

  // ---- stage 3: x2 = feat[:,60:180) -> feat[:,180:380) ----
  topk(feat + 60, 380, 120, ftop, 540);
  slice(ftop, 540, ia, 32, 0, 16, 1);  slice(ptop, 540, ia, 32, 16, 16, 1);
  slice(ftop, 540, ib, 16, 0, 6, 1);   slice(ptop, 540, ib, 16, 6, 6, 1);
  run_branch(feat + 60, 380, 120, E3b.a, 80, 80, ia, 32, 32, ea);
  run_branch(feat + 60, 380, 120, E3b.b, 80, 80, ib, 16, 12, eb);
  max2(80, 180);
  slice(ftop, 540, ia, 32, 0, 18, 30); slice(ptop, 540, ia, 32, 18, 14, 38);
  slice(ftop, 540, ib, 16, 0, 4, 135); slice(ptop, 540, ib, 16, 4, 6, 90);
  run_branch(feat + 60, 380, 120, D3b.a, 120, 120, ia, 32, 32, ea);
  run_branch(feat + 60, 380, 120, D3b.b, 120, 120, ib, 16, 10, eb);
  max2(120, 260);

  // ---- g path: g = feat_imp(lrelu(feat@g1)) -> cat[:, 0:400) ----
  launch_gemm(stream, feat, 380, G1.W, G1.b, bigA, 400, NPTS, 400, 380, 2);
  launch_gemm(stream, bigA, 400, GFI.W, GFI.b, bigB, 400, NPTS, 400, 400, 3);
  mul_kernel<<<(NPTS * 400 + 255) / 256, 256, 0, stream>>>(
      bigA, 400, bigB, 400, cat, 656, 0, 400, NPTS);

  // ---- grouped PointNet -> cat[:, 400:656) (stable group order) ----
  lut_kernel<<<(NPTS + 255) / 256, 256, 0, stream>>>(SEG, gid, NPTS);
  launch_gemm(stream, x0, 6, PN1.W, PN1.b, P, 64, NPTS, 64, 6, 1);
  launch_gemm(stream, P, 64, PN2.W, PN2.b, Q, 128, NPTS, 128, 64, 1);
  segmax_kernel<<<2, 256, 0, stream>>>(Q, gid, NPTS, g4);
  catgp_kernel<<<(NPTS * 256 + 255) / 256, 256, 0, stream>>>(Q, gid, g4, f256, NPTS);
  launch_gemm(stream, f256, 256, PN3.W, PN3.b, pn3, 256, NPTS, 256, 256, 1);
  rank_kernel<<<1, 1, 0, stream>>>(gid, dstof, NPTS);
  permscatter_kernel<<<(NPTS * 256 + 255) / 256, 256, 0, stream>>>(pn3, dstof, cat, NPTS);

  // ---- head MLPs ----
  launch_gemm(stream, cat, 656, MLP2.W, MLP2.b, bigA, 1024, NPTS, 1024, 656, 2);
  launch_gemm(stream, bigA, 1024, FI.W, FI.b, bigB, 1024, NPTS, 1024, 1024, 3);
  mul_kernel<<<(NPTS * 1024 + 255) / 256, 256, 0, stream>>>(
      bigA, 1024, bigB, 1024, bigA, 1024, 0, 1024, NPTS);
  // res block 1: 1024 -> 512
  launch_gemm(stream, bigA, 1024, RB1L1.W, RB1L1.b, r1, 512, NPTS, 512, 1024, 2);
  launch_gemm(stream, r1, 512, RB1L2.W, RB1L2.b, r2, 512, NPTS, 512, 512, 2);
  launch_gemm(stream, bigA, 1024, RB1RS.W, RB1RS.b, r3, 512, NPTS, 512, 1024, 2);
  add_kernel<<<(int)(((long)NPTS * 512 + 255) / 256), 256, 0, stream>>>(
      r2, r3, r1, (long)NPTS * 512);
  // res block 2: 512 -> 256
  launch_gemm(stream, r1, 512, RB2L1.W, RB2L1.b, bigB, 256, NPTS, 256, 512, 2);
  launch_gemm(stream, bigB, 256, RB2L2.W, RB2L2.b, bigB + (long)NPTS * 256, 256,
              NPTS, 256, 256, 2);
  launch_gemm(stream, r1, 512, RB2RS.W, RB2RS.b, r2, 256, NPTS, 256, 512, 2);
  add_kernel<<<(int)(((long)NPTS * 256 + 255) / 256), 256, 0, stream>>>(
      bigB + (long)NPTS * 256, r2, bigB, (long)NPTS * 256);
  // classifier
  launch_gemm(stream, bigB, 256, OUTL.W, OUTL.b, (float*)d_out, 17, NPTS, 17, 256, 0);
}
